// TDSSSMEncoder_56994216018370
// MI455X (gfx1250) — matrix operand order repaired and checked by reading, not compiled, not run on hardware
//
#include <hip/hip_runtime.h>
#include <math.h>

// ---------------------------------------------------------------------------
// Elman tanh SSM, restructured for CDNA5 (gfx1250, wave32, WMMA):
//   UB = u @ B^T                (big parallel bf16 WMMA GEMM -> ws)
//   d_out = u @ D^T             (big parallel bf16 WMMA GEMM -> d_out)
//   x_t = tanh(x_{t-1}@A^T+UB)  (persistent single-WG scan, A cached in LDS)
//   d_out += X @ C^T            (big parallel bf16 WMMA GEMM, accumulate)
// ---------------------------------------------------------------------------

typedef __attribute__((ext_vector_type(16))) __bf16 v16bf;
typedef __attribute__((ext_vector_type(8)))  __bf16 v8bf;
typedef __attribute__((ext_vector_type(8)))  float  v8f;

#define T_STEPS 4096
#define NB      32
#define F_DIM   512
#define TN      (T_STEPS * NB)          // 131072 rows
#define KT      (F_DIM / 32)            // 16 k-iterations per tile

__device__ __forceinline__ v8f wmma_bf16(v16bf a, v16bf b, v8f c) {
  // D = A(16x32) * B(32x16) + C, f32 accumulate
  return __builtin_amdgcn_wmma_f32_16x16x32_bf16(
      /*neg_a=*/false, a, /*neg_b=*/false, b,
      /*c_mod=*/(short)0, c, /*reuse_a=*/false, /*reuse_b=*/false);
}

// A-fragment (M=16 x K=32) from row-major bf16 [rows][ld].
// Contiguous k order (same as load_bfrag): lane<16 -> K 0..15 ; lane>=16 -> K 16..31, M=lane&15.
__device__ __forceinline__ v16bf load_afrag(const __bf16* __restrict__ p,
                                            int row0, int ld, int k0, int lane) {
  const int m  = row0 + (lane & 15);
  const int kb = k0 + ((lane >> 4) << 4);
  const __bf16* q = p + (size_t)m * ld + kb;
  union { v16bf v; v8bf h[2]; } u;
  u.h[0] = *(const v8bf*)(q);
  u.h[1] = *(const v8bf*)(q + 8);
  return u.v;
}

// B-fragment (K=32 x N=16), B(k,n) = W[col0+n][k] from row-major W[f_out][k].
// Contiguous k order (same as load_afrag): lanes 0-15 hold K=0..15, lanes 16-31 hold K=16..31, N=lane&15.
__device__ __forceinline__ v16bf load_bfrag(const __bf16* __restrict__ W,
                                            int col0, int ld, int k0, int lane) {
  const int n  = col0 + (lane & 15);
  const int kk = k0 + ((lane >> 4) << 4);
  return *(const v16bf*)(W + (size_t)n * ld + kk);
}

// ---------------------------------------------------------------------------
// Generic  O[TN x 512] (+)= S[TN x 512] @ W^T   (bf16 in, f32 out)
// 256 threads = 8 waves; each wave: 2 M-tiles x 4 N-tiles (A and B frag
// reuse; 8 WMMAs per K-step). Next K-step's B-frags are issued before the
// current WMMAs and pinned there with a scheduling fence so loads overlap
// matrix math instead of serializing on s_wait_loadcnt 0.
// grid = 4096 blocks covers 4096 M-pairs x 8 N-tile-groups exactly.
// ---------------------------------------------------------------------------
__global__ __launch_bounds__(256) void gemm512_bf16(
    const __bf16* __restrict__ S, const __bf16* __restrict__ W,
    float* __restrict__ O, int accumulate) {
  const int lane = threadIdx.x & 31;
  const int wave = threadIdx.x >> 5;
  const int gw   = blockIdx.x * 8 + wave;
  const int mp   = gw >> 3;          // M tile pair (32 rows)
  const int ng   = gw & 7;           // N tile group (4 x 16 cols)
  const int r0   = mp << 5;
  const int hi8  = (lane >> 4) << 3;
  const int cl   = lane & 15;

  v8f acc[2][4];
  if (accumulate) {
#pragma unroll
    for (int m = 0; m < 2; ++m)
#pragma unroll
      for (int j = 0; j < 4; ++j) {
        const int c0 = (ng * 4 + j) << 4;
#pragma unroll
        for (int i = 0; i < 8; ++i)
          acc[m][j][i] = O[(size_t)(r0 + 16 * m + hi8 + i) * F_DIM + c0 + cl];
      }
  } else {
#pragma unroll
    for (int m = 0; m < 2; ++m)
#pragma unroll
      for (int j = 0; j < 4; ++j)
#pragma unroll
        for (int i = 0; i < 8; ++i) acc[m][j][i] = 0.0f;
  }

  // prologue: B-fragments for kt = 0
  v16bf b[4];
#pragma unroll
  for (int j = 0; j < 4; ++j) b[j] = load_bfrag(W, (ng * 4 + j) << 4, F_DIM, 0, lane);

#pragma unroll
  for (int kt = 0; kt < KT; ++kt) {
    const int k0 = kt * 32;
    // A-fragments for this K-step
    v16bf a0 = load_afrag(S, r0,      F_DIM, k0, lane);
    v16bf a1 = load_afrag(S, r0 + 16, F_DIM, k0, lane);
    // B-fragments for the next K-step, issued ahead of the WMMAs
    v16bf bn[4];
    if (kt + 1 < KT) {
#pragma unroll
      for (int j = 0; j < 4; ++j)
        bn[j] = load_bfrag(W, (ng * 4 + j) << 4, F_DIM, k0 + 32, lane);
    }
    __builtin_amdgcn_sched_barrier(0);   // keep the loads above the math
#pragma unroll
    for (int j = 0; j < 4; ++j) {
      acc[0][j] = wmma_bf16(a0, b[j], acc[0][j]);
      acc[1][j] = wmma_bf16(a1, b[j], acc[1][j]);
    }
    if (kt + 1 < KT) {
#pragma unroll
      for (int j = 0; j < 4; ++j) b[j] = bn[j];
    }
  }

#pragma unroll
  for (int m = 0; m < 2; ++m)
#pragma unroll
    for (int j = 0; j < 4; ++j) {
      const int c0 = (ng * 4 + j) << 4;
#pragma unroll
      for (int i = 0; i < 8; ++i)
        O[(size_t)(r0 + 16 * m + hi8 + i) * F_DIM + c0 + cl] = acc[m][j][i];
    }
}

// ---------------------------------------------------------------------------
// Sequential scan: one persistent workgroup, 1024 threads = 32 waves.
// Wave w owns output columns [16w, 16w+16). State x lives in LDS (padded,
// bank-conflict-free b128 reads). Waves 0..15 keep their A column-slice
// resident in LDS for all 4096 steps; waves 16..31 stream theirs from L2.
// Each tile uses TWO independent accumulator chains (even/odd K pairs) to
// halve the dependent-WMMA critical path of every sequential step.
// ---------------------------------------------------------------------------
#define XS_STRIDE 520     // 512 + 8 bf16 pad: row step = 4 banks -> no conflicts
#define AS_STRIDE 528     // 512 + 16 bf16 pad for 32B B-frag reads
#define NCACHE    16      // column-slices of A cached in LDS (16 * 16.5KB)

__global__ __launch_bounds__(1024) void scan_tanh(
    const __bf16* __restrict__ Abf, const float* __restrict__ UB,
    __bf16* __restrict__ Xbf) {
  __shared__ __align__(32) __bf16 xs[NB][XS_STRIDE];            // ~33 KB
  __shared__ __align__(32) __bf16 As[NCACHE][16 * AS_STRIDE];   // ~270 KB

  const int lane = threadIdx.x & 31;
  const int wave = threadIdx.x >> 5;   // 0..31 -> column tile
  const int c0   = wave << 4;
  const int hi   = lane >> 4;
  const int hi8  = hi << 3;
  const int cl   = lane & 15;

  // x0 = 0
  for (int i = threadIdx.x; i < NB * XS_STRIDE; i += 1024)
    xs[i / XS_STRIDE][i % XS_STRIDE] = (__bf16)0.0f;
  // cache A slices (rows c0..c0+15, all K) fragment-ready in LDS
  if (wave < NCACHE) {
    for (int i = lane; i < 16 * (F_DIM / 8); i += 32) {
      const int col = i >> 6;            // 64 8-elt chunks per row
      const int k8  = (i & 63) << 3;
      *(v8bf*)&As[wave][col * AS_STRIDE + k8] =
          *(const v8bf*)&Abf[(size_t)(c0 + col) * F_DIM + k8];
    }
  }
  __syncthreads();

  for (int t = 0; t < T_STEPS; ++t) {
    const float* __restrict__ ubt = UB + (size_t)t * NB * F_DIM;
    if (t + 1 < T_STEPS)   // pull next step's drive term toward the WGP
      __builtin_prefetch(ubt + NB * F_DIM + threadIdx.x * 16, 0, 0);

    v8f acc0a, acc0b, acc1a, acc1b;
#pragma unroll
    for (int i = 0; i < 8; ++i) {
      acc0a[i] = 0.0f; acc0b[i] = 0.0f; acc1a[i] = 0.0f; acc1b[i] = 0.0f;
    }

#pragma unroll
    for (int kt = 0; kt < KT; kt += 2) {
      const int k0 = kt * 32;
      const int k1 = k0 + 32;
      // issue all fragment loads for the K-pair up front
      v16bf bA, bB;
      if (wave < NCACHE) {
        bA = *(const v16bf*)&As[wave][cl * AS_STRIDE + k0 + (hi << 4)];
        bB = *(const v16bf*)&As[wave][cl * AS_STRIDE + k1 + (hi << 4)];
      } else {
        const __bf16* base = Abf + (size_t)(c0 + cl) * F_DIM + (hi << 4);
        bA = *(const v16bf*)(base + k0);
        bB = *(const v16bf*)(base + k1);
      }
      union { v16bf v; v8bf h[2]; } a00, a10, a01, a11;
      const int kb0 = k0 + (hi << 4);
      const int kb1 = k1 + (hi << 4);
      a00.h[0] = *(const v8bf*)&xs[cl][kb0];
      a00.h[1] = *(const v8bf*)&xs[cl][kb0 + 8];
      a10.h[0] = *(const v8bf*)&xs[16 + cl][kb0];
      a10.h[1] = *(const v8bf*)&xs[16 + cl][kb0 + 8];
      a01.h[0] = *(const v8bf*)&xs[cl][kb1];
      a01.h[1] = *(const v8bf*)&xs[cl][kb1 + 8];
      a11.h[0] = *(const v8bf*)&xs[16 + cl][kb1];
      a11.h[1] = *(const v8bf*)&xs[16 + cl][kb1 + 8];
      // two independent chains per output tile
      acc0a = wmma_bf16(a00.v, bA, acc0a);
      acc1a = wmma_bf16(a10.v, bA, acc1a);
      acc0b = wmma_bf16(a01.v, bB, acc0b);
      acc1b = wmma_bf16(a11.v, bB, acc1b);
    }
    __syncthreads();   // all waves done reading xs

    __bf16* __restrict__ xo = Xbf + (size_t)t * NB * F_DIM;
#pragma unroll
    for (int i = 0; i < 8; ++i) {
      const int r = hi8 + i;
      const int c = c0 + cl;
      const float v0 = tanhf(acc0a[i] + acc0b[i] + ubt[r * F_DIM + c]);
      const float v1 = tanhf(acc1a[i] + acc1b[i] + ubt[(16 + r) * F_DIM + c]);
      const __bf16 b0 = (__bf16)v0, b1 = (__bf16)v1;
      xs[r][c]      = b0;
      xs[16 + r][c] = b1;
      xo[r * F_DIM + c]        = b0;
      xo[(16 + r) * F_DIM + c] = b1;
    }
    __syncthreads();   // new state visible to every wave
  }
}

__global__ void cvt_f32_bf16(const float* __restrict__ in,
                             __bf16* __restrict__ out, long n) {
  long i = (long)blockIdx.x * blockDim.x + threadIdx.x;
  const long st = (long)gridDim.x * blockDim.x;
  for (; i < n; i += st) out[i] = (__bf16)in[i];
}

extern "C" void kernel_launch(void* const* d_in, const int* in_sizes, int n_in,
                              void* d_out, int out_size, void* d_ws, size_t ws_size,
                              hipStream_t stream) {
  (void)in_sizes; (void)n_in; (void)out_size; (void)ws_size;
  const float* u = (const float*)d_in[0];
  const float* A = (const float*)d_in[1];
  const float* B = (const float*)d_in[2];
  const float* C = (const float*)d_in[3];
  const float* D = (const float*)d_in[4];
  float* out = (float*)d_out;

  const size_t MAT = (size_t)F_DIM * F_DIM;   // 262144
  const size_t TNF = (size_t)TN * F_DIM;      // 67108864

  char* ws = (char*)d_ws;
  size_t off = 0;
  __bf16* Abf = (__bf16*)(ws + off); off += MAT * 2;
  __bf16* Bbf = (__bf16*)(ws + off); off += MAT * 2;
  __bf16* Cbf = (__bf16*)(ws + off); off += MAT * 2;
  __bf16* Dbf = (__bf16*)(ws + off); off += MAT * 2;
  __bf16* Ubf = (__bf16*)(ws + off); off += TNF * 2;   // u in bf16
  __bf16* Xbf = (__bf16*)(ws + off); off += TNF * 2;   // all x_t in bf16
  float*  UBt = (float*)(ws + off);  off += TNF * 4;   // u @ B^T in f32

  // precision downconverts
  cvt_f32_bf16<<<8192, 256, 0, stream>>>(u, Ubf, (long)TNF);
  cvt_f32_bf16<<<1024, 256, 0, stream>>>(A, Abf, (long)MAT);
  cvt_f32_bf16<<<1024, 256, 0, stream>>>(B, Bbf, (long)MAT);
  cvt_f32_bf16<<<1024, 256, 0, stream>>>(C, Cbf, (long)MAT);
  cvt_f32_bf16<<<1024, 256, 0, stream>>>(D, Dbf, (long)MAT);

  // parallel drive terms
  gemm512_bf16<<<4096, 256, 0, stream>>>(Ubf, Bbf, UBt, 0);   // UB = u B^T
  gemm512_bf16<<<4096, 256, 0, stream>>>(Ubf, Dbf, out, 0);   // out = u D^T

  // sequential recurrence (persistent single workgroup)
  scan_tanh<<<1, 1024, 0, stream>>>(Abf, UBt, Xbf);

  // out += X C^T
  gemm512_bf16<<<4096, 256, 0, stream>>>(Xbf, Cbf, out, 1);
}
